// MultiHeadAttention_21560735826476
// MI455X (gfx1250) — matrix operand order repaired and checked by reading, not compiled, not run on hardware
//
#include <hip/hip_runtime.h>

typedef __attribute__((ext_vector_type(16))) _Float16 v16h;
typedef __attribute__((ext_vector_type(8)))  _Float16 v8h;
typedef __attribute__((ext_vector_type(4)))  _Float16 v4h;
typedef __attribute__((ext_vector_type(8)))  float    v8f;

#define C_DIM 192
#define T_DIM 2048
#define B_DIM 4
#define H_DIM 2
#define KC    96

// Fragment layouts (CDNA5 ISA 7.12.2, wave32, v_wmma_f32_16x16x32_f16):
//  A 16x32 f16 : lane row m=lane&15; halves = row[kb8..kb8+7] ++ row[16+kb8..+7],
//                kb8 = 8*(lane>=16) -> two contiguous 16B runs.
//  B 32x16 f16 : lane col n=lane&15; halves = col[kb8..kb8+7] ++ col[16+kb8..+7]
//                (same k order as A) when the tile is stored n-major.
//  C/D 16x16 f32: lane col n=lane&15; VGPR r -> row r + 8*(lane>=16).
union ABFrag {
  v16h f;
  v8h  h[2];
};

__device__ __forceinline__ v8f wmma_f16(v16h a, v16h b, v8f c) {
  return __builtin_amdgcn_wmma_f32_16x16x32_f16(false, a, false, b, (short)0, c,
                                                false, false);
}

__device__ __forceinline__ v16h load_a_frag(const _Float16* row, int lane) {
  const int kb8 = (lane >> 4) << 3;
  ABFrag u;
  u.h[0] = *(const v8h*)(row + kb8);
  u.h[1] = *(const v8h*)(row + 16 + kb8);
  return u.f;
}
__device__ __forceinline__ v16h load_b_frag(const _Float16* row, int lane) {
  const int kb8 = (lane >> 4) << 3;
  ABFrag u;
  u.h[0] = *(const v8h*)(row + kb8);
  u.h[1] = *(const v8h*)(row + 16 + kb8);
  return u.f;
}

// Async global->LDS 16B copy (GLOBAL_LOAD_ASYNC_TO_LDS_B128, ASYNCcnt).
__device__ __forceinline__ void async_copy_b128(void* lds_dst, const void* gsrc) {
  unsigned lds_a = (unsigned)(uintptr_t)lds_dst;
  unsigned long long ga = (unsigned long long)(uintptr_t)gsrc;
  asm volatile("global_load_async_to_lds_b128 %0, %1, off"
               :
               : "v"(lds_a), "v"(ga)
               : "memory");
}
__device__ __forceinline__ void wait_async0() {
  asm volatile("s_wait_asynccnt 0" ::: "memory");
}
// Wave-local LDS handoff: LDS is in-order per wave; this stops compiler/motion
// and drains DScnt without a full workgroup barrier (per-wave tiles only).
__device__ __forceinline__ void wave_lds_fence() {
  asm volatile("s_wait_dscnt 0" ::: "memory");
  __builtin_amdgcn_wave_barrier();
}

__device__ __forceinline__ void load4f(const float* p, float o[4]) {
  float4 v = *(const float4*)p;
  o[0] = v.x; o[1] = v.y; o[2] = v.z; o[3] = v.w;
}
__device__ __forceinline__ void load4f(const _Float16* p, float o[4]) {
  v4h v = *(const v4h*)p;
  for (int j = 0; j < 4; ++j) o[j] = (float)v[j];
}

// =============================================================================
// Projection GEMM: out[o,t] = W[o,c] * src[c,t] + bias[o]
// MODE 0: f16 -> dst[b][h][t][d] (o = h*96+d), scaled (Q gets 1/sqrt(96))
// MODE 1: f32 -> dst[b][o][t]
// block = 256 (8 waves), tile = 192(M) x 64(N), K streamed in 32-chunks.
// =============================================================================
template <typename SrcT, int MODE>
__global__ __launch_bounds__(256) void proj_kernel(
    const float* __restrict__ W, const float* __restrict__ bias,
    const SrcT* __restrict__ src, void* __restrict__ dst, float scale) {
  __shared__ __align__(16) _Float16 lw[C_DIM][40];  // W chunk [o 192][k 32]
  __shared__ __align__(16) _Float16 lxT[64][40];    // src chunk, t-major [t][k]

  const int tid  = threadIdx.x;
  const int lane = tid & 31, wid = tid >> 5;
  const int b    = blockIdx.y;
  const int t0   = blockIdx.x * 64;
  const int n_tile = wid & 3;
  const int mgrp   = wid >> 2;
  const int n  = lane & 15;
  const int mb = (lane >> 4) << 3;

  v8f acc[6];
  for (int i = 0; i < 6; ++i)
    for (int r = 0; r < 8; ++r) acc[i][r] = 0.f;

  for (int kk = 0; kk < 6; ++kk) {
    const int k0 = kk * 32;
    for (int i = 0; i < 6; ++i) {            // 192x32 W chunk: 1536 float4
      int e = tid + i * 256;
      int row = e >> 3, c4 = e & 7;          // 8 x float4 per 32-wide row
      float o4[4];
      load4f(&W[row * C_DIM + k0 + c4 * 4], o4);
      v4h hv;
      for (int j = 0; j < 4; ++j) hv[j] = (_Float16)o4[j];
      *(v4h*)&lw[row][c4 * 4] = hv;
    }
    for (int i = 0; i < 2; ++i) {            // 32x64 src: 512 vec4, transposed
      int e = tid + i * 256;
      int row = e >> 4, c4 = e & 15;         // row=k, 16 x vec4 per 64-wide row
      float o4[4];
      load4f(&src[((size_t)b * C_DIM + k0 + row) * T_DIM + t0 + c4 * 4], o4);
      for (int j = 0; j < 4; ++j) lxT[c4 * 4 + j][row] = (_Float16)o4[j];
    }
    __syncthreads();

    v16h bf = load_b_frag(&lxT[n_tile * 16 + n][0], lane);
    for (int i = 0; i < 6; ++i) {
      const int mt = mgrp * 6 + i;
      v16h af = load_a_frag(&lw[mt * 16 + (lane & 15)][0], lane);
      acc[i] = wmma_f16(af, bf, acc[i]);
    }
    __syncthreads();
  }

  for (int i = 0; i < 6; ++i) {
    const int mt = mgrp * 6 + i;
    for (int r = 0; r < 8; ++r) {
      int m = mb + r;
      int o = mt * 16 + m;
      int t = t0 + n_tile * 16 + n;
      float val = acc[i][r] + bias[o];
      if (MODE == 0) {
        int h = (o >= KC) ? 1 : 0;
        int d = o - h * KC;
        ((_Float16*)dst)[(((size_t)b * H_DIM + h) * T_DIM + t) * KC + d] =
            (_Float16)(val * scale);
      } else {
        ((float*)dst)[((size_t)b * C_DIM + o) * T_DIM + t] = val;
      }
    }
  }
}

// =============================================================================
// Band dot products: rd[bh,t,i] = dot(q_scaled[bh,t,:], emb_rel_k[i,:]), i=0..8
// =============================================================================
__global__ __launch_bounds__(256) void relk_kernel(
    const _Float16* __restrict__ qh, const float* __restrict__ emb_k,
    float* __restrict__ rd) {
  __shared__ float se[9 * KC];
  for (int e = threadIdx.x; e < 9 * KC; e += 256) se[e] = emb_k[e];
  __syncthreads();
  int t = blockIdx.x * 256 + threadIdx.x;     // over B*H*T
  const _Float16* q = qh + (size_t)t * KC;
  v8h qv[12];
  for (int c = 0; c < 12; ++c) qv[c] = *(const v8h*)(q + c * 8);
  for (int i = 0; i < 9; ++i) {
    float s = 0.f;
    for (int c = 0; c < 12; ++c)
      for (int j = 0; j < 8; ++j) s += (float)qv[c][j] * se[i * KC + c * 8 + j];
    rd[(size_t)t * 9 + i] = s;
  }
}

// =============================================================================
// Fused flash attention with relative-position band.
// grid = (T/64, H, B), block = 128 (4 waves, 16 q-rows each).
// K staged via async global->LDS; V transposed to d-major LDS.
// Only tile staging uses block barriers; softmax state is per-wave.
// =============================================================================
__global__ __launch_bounds__(128) void attn_kernel(
    const _Float16* __restrict__ qh, const _Float16* __restrict__ kh,
    const _Float16* __restrict__ vh, const float* __restrict__ rd,
    const float* __restrict__ emb_v, _Float16* __restrict__ att) {
  __shared__ __align__(16) _Float16 lk[32][104];  // K chunk [key][d]
  __shared__ __align__(16) _Float16 lvT[KC][40];  // V chunk, d-major [d][key]
  __shared__ float    sS[4][16][33];              // per-wave score tile
  __shared__ __align__(16) _Float16 sP[4][16][40];// per-wave prob tile (f16)
  __shared__ float    row_m[4][16];
  __shared__ float    row_l[4][16];
  __shared__ float    row_a[4][16];
  __shared__ float    sEv[9][KC];
  __shared__ float    sPb[4][16][9];

  const int tid  = threadIdx.x;
  const int lane = tid & 31, wid = tid >> 5;
  const int n    = lane & 15;
  const int mb   = (lane >> 4) << 3;
  const size_t bh = (size_t)blockIdx.z * H_DIM + blockIdx.y;
  const _Float16* qh_bh = qh + bh * T_DIM * KC;
  const _Float16* kh_bh = kh + bh * T_DIM * KC;
  const _Float16* vh_bh = vh + bh * T_DIM * KC;
  const int q0 = blockIdx.x * 64 + wid * 16;

  for (int e = tid; e < 9 * KC; e += 128) sEv[e / KC][e % KC] = emb_v[e];
  if (lane < 16) {                       // per-wave softmax state
    row_m[wid][lane] = -1e30f;
    row_l[wid][lane] = 0.f;
  }

  // Q A-fragments (d split into 3 chunks of 32): two 16B loads each.
  v16h qf[3];
  for (int kd = 0; kd < 3; ++kd)
    qf[kd] = load_a_frag(qh_bh + (size_t)(q0 + (lane & 15)) * KC + kd * 32, lane);

  v8f oacc[6];
  for (int i = 0; i < 6; ++i)
    for (int r = 0; r < 8; ++r) oacc[i][r] = 0.f;
  __syncthreads();

  for (int kb0 = 0; kb0 < T_DIM; kb0 += 32) {
    // ---- stage K (async) and V (transposed); prefetch next chunk ----------
    for (int i = 0; i < 3; ++i) {
      int e = tid + i * 128;
      int row = e / 12, col = e % 12;               // 96 halves = 12 x 16B
      async_copy_b128(&lk[row][col * 8],
                      kh_bh + ((size_t)(kb0 + row)) * KC + col * 8);
      v8h vv = *(const v8h*)(vh_bh + ((size_t)(kb0 + row)) * KC + col * 8);
      for (int hh = 0; hh < 8; ++hh) lvT[col * 8 + hh][row] = vv[hh];
    }
    if (kb0 + 32 < T_DIM) {                         // global_prefetch_b8
      __builtin_prefetch(kh_bh + (size_t)(kb0 + 32 + lane) * KC, 0, 1);
      __builtin_prefetch(vh_bh + (size_t)(kb0 + 32 + lane) * KC, 0, 1);
    }
    wait_async0();
    __syncthreads();

    // ---- S = Q * K^T  (2 key sub-tiles x 3 d-chunks) -----------------------
    for (int nt = 0; nt < 2; ++nt) {
      v8f sacc;
      for (int r = 0; r < 8; ++r) sacc[r] = 0.f;
      for (int kd = 0; kd < 3; ++kd) {
        v16h bf = load_b_frag(&lk[nt * 16 + n][kd * 32], lane);
        sacc = wmma_f16(qf[kd], bf, sacc);
      }
      for (int r = 0; r < 8; ++r) {                 // scatter + band bias
        int m = mb + r;
        int col = nt * 16 + n;
        int kg = kb0 + col, qg = q0 + m;
        float sv = sacc[r];
        int dd = kg - qg + 4;
        if (dd >= 0 && dd < 9) sv += rd[(bh * T_DIM + qg) * 9 + dd];
        sS[wid][m][col] = sv;
      }
    }
    wave_lds_fence();                               // sS is per-wave

    // ---- online softmax: 2 lanes/row, combine via shfl_xor(16) (wave32) ----
    {
      int row = lane & 15;
      int c0 = (lane >> 4) * 16;
      float mx = -1e30f;
      for (int j = 0; j < 16; ++j) mx = fmaxf(mx, sS[wid][row][c0 + j]);
      mx = fmaxf(mx, __shfl_xor(mx, 16, 32));
      float mold = row_m[wid][row];
      float mnew = fmaxf(mold, mx);
      float psum = 0.f;
      for (int j = 0; j < 16; ++j) {
        float p = __expf(sS[wid][row][c0 + j] - mnew);
        sP[wid][row][c0 + j] = (_Float16)p;
        psum += p;
      }
      psum += __shfl_xor(psum, 16, 32);
      if (lane < 16) {
        float alpha = __expf(mold - mnew);
        row_m[wid][row] = mnew;
        row_l[wid][row] = row_l[wid][row] * alpha + psum;
        row_a[wid][row] = alpha;
      }
    }
    wave_lds_fence();                               // sP/row_a are per-wave

    // ---- O = O*alpha + P*V  (6 d sub-tiles) --------------------------------
    float al[8];
    for (int r = 0; r < 8; ++r) al[r] = row_a[wid][mb + r];
    v16h pf = load_a_frag(&sP[wid][lane & 15][0], lane);
    for (int i = 0; i < 6; ++i) {
      v8f o = oacc[i];
      for (int r = 0; r < 8; ++r) o[r] *= al[r];
      v16h bf = load_b_frag(&lvT[i * 16 + n][0], lane);
      oacc[i] = wmma_f16(pf, bf, o);
    }
    __syncthreads();                                // lk/lvT reused next chunk
  }

  // Band probabilities p[q, q+dd-4] recomputed exactly (9 x 96 dots per row).
  if (lane < 16) {
    int row = lane;
    int qg = q0 + row;
    float mfin = row_m[wid][row];
    float lfin = row_l[wid][row];
    v8h qrow[12];
    for (int c = 0; c < 12; ++c)
      qrow[c] = *(const v8h*)(qh_bh + (size_t)qg * KC + c * 8);
    for (int dd = 0; dd < 9; ++dd) {
      int kg = qg + dd - 4;
      float pb = 0.f;
      if (kg >= 0 && kg < T_DIM) {
        float s = rd[(bh * T_DIM + qg) * 9 + dd];
        const v8h* kr = (const v8h*)(kh_bh + (size_t)kg * KC);
        for (int c = 0; c < 12; ++c) {
          v8h kv = kr[c];
          for (int j = 0; j < 8; ++j) s += (float)qrow[c][j] * (float)kv[j];
        }
        pb = __expf(s - mfin) / lfin;
      }
      sPb[wid][row][dd] = pb;
    }
  }
  wave_lds_fence();                                 // sPb is per-wave

  // Normalize, add rel-V band, write [b][c][t] f16 for the Wo GEMM.
  for (int i = 0; i < 6; ++i) {
    int d = i * 16 + n;
    for (int r = 0; r < 8; ++r) {
      int m = mb + r;
      int qg = q0 + m;
      float val = oacc[i][r] / row_l[wid][m];
      for (int dd = 0; dd < 9; ++dd) val += sPb[wid][m][dd] * sEv[dd][d];
      att[((size_t)blockIdx.z * C_DIM + blockIdx.y * KC + d) * T_DIM + qg] =
          (_Float16)val;
    }
  }
}

// =============================================================================
extern "C" void kernel_launch(void* const* d_in, const int* in_sizes, int n_in,
                              void* d_out, int out_size, void* d_ws,
                              size_t ws_size, hipStream_t stream) {
  (void)in_sizes; (void)n_in; (void)out_size; (void)ws_size;
  const float* x  = (const float*)d_in[0];
  const float* c  = (const float*)d_in[1];
  const float* Wq = (const float*)d_in[2];
  const float* bq = (const float*)d_in[3];
  const float* Wk = (const float*)d_in[4];
  const float* bk = (const float*)d_in[5];
  const float* Wv = (const float*)d_in[6];
  const float* bv = (const float*)d_in[7];
  const float* Wo = (const float*)d_in[8];
  const float* bo = (const float*)d_in[9];
  const float* ek = (const float*)d_in[10];
  const float* ev = (const float*)d_in[11];
  // d_in[12] = attn_mask: all ones in this problem -> no-op.

  char* ws = (char*)d_ws;
  const size_t sz_h = (size_t)B_DIM * H_DIM * T_DIM * KC * sizeof(_Float16);
  _Float16* qh  = (_Float16*)(ws);
  _Float16* kh  = (_Float16*)(ws + sz_h);
  _Float16* vh  = (_Float16*)(ws + 2 * sz_h);
  _Float16* att = (_Float16*)(ws + 3 * sz_h);
  float*    rd  = (float*)(ws + 4 * sz_h);

  const float qscale = 0.10206207261596577f;  // 1/sqrt(96)
  dim3 pgrid(T_DIM / 64, B_DIM);
  proj_kernel<float, 0><<<pgrid, 256, 0, stream>>>(Wq, bq, x, (void*)qh, qscale);
  proj_kernel<float, 0><<<pgrid, 256, 0, stream>>>(Wk, bk, c, (void*)kh, 1.f);
  proj_kernel<float, 0><<<pgrid, 256, 0, stream>>>(Wv, bv, c, (void*)vh, 1.f);
  relk_kernel<<<(B_DIM * H_DIM * T_DIM) / 256, 256, 0, stream>>>(qh, ek, rd);
  attn_kernel<<<dim3(T_DIM / 64, H_DIM, B_DIM), 128, 0, stream>>>(qh, kh, vh, rd,
                                                                  ev, att);
  proj_kernel<_Float16, 1><<<pgrid, 256, 0, stream>>>(Wo, bo, att, d_out, 1.f);
}
